// Hydra_64896955842652
// MI455X (gfx1250) — compile-verified, matrix operand order checked by reading, not run on hardware
//
#include <hip/hip_runtime.h>

// ---------------------------------------------------------------------------
// Hydra (bidirectional Mamba2) layer for MI455X / gfx1250.
// B=2, L=4096, D_MODEL=1024, D_INNER=2048, NHEADS=32, HEADDIM=64,
// D_STATE=64, NGROUPS=1, D_XBC=2304, D_IN_PROJ=4416, D_CONV=7.
// ---------------------------------------------------------------------------

typedef __attribute__((ext_vector_type(16))) __bf16 v16bf;
typedef __attribute__((ext_vector_type(8)))  __bf16 v8bf;
typedef __attribute__((ext_vector_type(8)))  float  v8f;

#define LQ 4096
#define DMODEL 1024
#define DINNER 2048
#define NHEADS 32
#define HEADDIM 64
#define DSTATE 64
#define DXBC 2304
#define DINPROJ 4416
#define PITCH1 4480   // in_proj output pitch padded to a multiple of 128
#define MROWS 8192    // 2 * 4096

__device__ __forceinline__ unsigned short bf16_rne(float f) {
    unsigned u = __float_as_uint(f);
    unsigned r = u + 0x7FFFu + ((u >> 16) & 1u);
    return (unsigned short)(r >> 16);
}

__device__ __forceinline__ float silu_f(float v) {
    return v / (1.0f + __expf(-v));
}

// gfx1250 async copy: 16 bytes global -> LDS, tracked by ASYNCcnt.
// LDS operand is the 32-bit LDS byte offset (generic addr low 32 bits).
__device__ __forceinline__ void async_load16(void* lds_dst, const void* gsrc) {
    unsigned ldsoff = (unsigned)(unsigned long long)lds_dst;
    asm volatile("global_load_async_to_lds_b128 %0, %1, off"
                 :: "v"(ldsoff), "v"(gsrc) : "memory");
}
__device__ __forceinline__ void async_wait0() {
    asm volatile("s_wait_asynccnt 0x0" ::: "memory");
}

// ---------------------------------------------------------------------------
// f32 -> bf16 (RNE) conversion
// ---------------------------------------------------------------------------
__global__ __launch_bounds__(256) void cvt_bf16_kernel(
    const float* __restrict__ in, unsigned short* __restrict__ out, int n)
{
    int i = blockIdx.x * 256 + threadIdx.x;
    if (i < n) out[i] = bf16_rne(in[i]);
}

// ---------------------------------------------------------------------------
// bf16 GEMM: Out[M,N] = A[M,K] * W[N,K]^T, f32 accumulate via WMMA.
// Block: 256 threads (8 waves). Block tile 256(M) x 128(N).
// Waves: 4 along M x 2 along N -> wave tile 64x64 -> 4x4 wmma frags
// (16 wmma per 16 ds_load_b128 per K-step of 32).
// Tile staging via GLOBAL_LOAD_ASYNC_TO_LDS_B128 (ASYNCcnt).
// ---------------------------------------------------------------------------
__global__ __launch_bounds__(256) void gemm_bf16_wmma(
    const __bf16* __restrict__ A, const __bf16* __restrict__ W,
    float* __restrict__ Out, int K, int N)
{
    __shared__ __align__(16) __bf16 ldsA[256 * 40];  // 32-col rows padded to 40
    __shared__ __align__(16) __bf16 ldsB[128 * 40];

    const int tid  = threadIdx.x;
    const int wave = tid >> 5;
    const int lane = tid & 31;
    const int wm = wave >> 1;        // 0..3 (M)
    const int wn = wave & 1;         // 0..1 (N)
    const int lhalf = lane >> 4;     // 0/1
    const int lrow  = lane & 15;     // 0..15
    const int m0 = blockIdx.y * 256;
    const int n0 = blockIdx.x * 128;

    // B staging: 128 rows x 32 cols, 16 elems/thread
    const int rb = tid >> 1;
    const int cb = (tid & 1) << 4;

    v8f acc[4][4] = {};

    for (int k0 = 0; k0 < K; k0 += 32) {
        __syncthreads();
        {
            // A: one full 32-elem row per thread (4 x 16B async)
            const __bf16* pa = A + (size_t)(m0 + tid) * K + k0;
            __bf16* la = &ldsA[tid * 40];
            async_load16(la,      pa);
            async_load16(la + 8,  pa + 8);
            async_load16(la + 16, pa + 16);
            async_load16(la + 24, pa + 24);
            // B: half row per thread (2 x 16B async)
            const __bf16* pb = W + (size_t)(n0 + rb) * K + k0 + cb;
            __bf16* lb = &ldsB[rb * 40 + cb];
            async_load16(lb,     pb);
            async_load16(lb + 8, pb + 8);
            if (k0 + 32 < K) {               // gfx1250: global_prefetch_b8
                __builtin_prefetch(pa + 32, 0, 3);
                __builtin_prefetch(pb + 32, 0, 3);
            }
        }
        async_wait0();
        __syncthreads();

        // ISA 16-bit A layout: lane half 0 covers K {0..7,16..23},
        // half 1 covers K {8..15,24..31} -> two b128 LDS reads per frag.
        union Frag { v16bf v; v8bf h[2]; };
        Frag fa[4], fb[4];
        const int kc = lhalf * 8;
#pragma unroll
        for (int i = 0; i < 4; ++i) {
            int row = wm * 64 + i * 16 + lrow;
            fa[i].h[0] = *(const v8bf*)&ldsA[row * 40 + kc];
            fa[i].h[1] = *(const v8bf*)&ldsA[row * 40 + kc + 16];
        }
#pragma unroll
        for (int j = 0; j < 4; ++j) {
            int row = wn * 64 + j * 16 + lrow;
            fb[j].h[0] = *(const v8bf*)&ldsB[row * 40 + kc];
            fb[j].h[1] = *(const v8bf*)&ldsB[row * 40 + kc + 16];
        }
#pragma unroll
        for (int i = 0; i < 4; ++i)
#pragma unroll
            for (int j = 0; j < 4; ++j)
                acc[i][j] = __builtin_amdgcn_wmma_f32_16x16x32_bf16(
                    false, fa[i].v, false, fb[j].v,
                    (short)0, acc[i][j], false, false);
    }

    // f32 C/D layout: VGPR r -> M = r + 8*lhalf, N = lane%16
#pragma unroll
    for (int i = 0; i < 4; ++i)
#pragma unroll
        for (int j = 0; j < 4; ++j) {
            int mbase = m0 + wm * 64 + i * 16 + lhalf * 8;
            int n = n0 + wn * 64 + j * 16 + lrow;
#pragma unroll
            for (int r = 0; r < 8; ++r)
                Out[(size_t)(mbase + r) * N + n] = acc[i][j][r];
        }
}

// ---------------------------------------------------------------------------
// Depthwise conv1d (k=7, SAME) + SiLU over the xBC slice of zxBCdt.
// ---------------------------------------------------------------------------
__global__ __launch_bounds__(256) void conv_silu_kernel(
    const float* __restrict__ zxBCdt, const float* __restrict__ cw,
    const float* __restrict__ cbias, float* __restrict__ xBC)
{
    size_t idx = (size_t)blockIdx.x * 256 + threadIdx.x;
    int c = (int)(idx % DXBC);
    int l = (int)((idx / DXBC) % LQ);
    int b = (int)(idx / ((size_t)DXBC * LQ));
    float acc = cbias[c];
#pragma unroll
    for (int j = 0; j < 7; ++j) {
        int lj = l + j - 3;
        if (lj >= 0 && lj < LQ)
            acc += zxBCdt[((size_t)(b * LQ + lj)) * PITCH1 + DINNER + c] * cw[c * 7 + j];
    }
    xBC[idx] = silu_f(acc);
}

// ---------------------------------------------------------------------------
// Bidirectional selective scan. One block per (dir*batch, head) = 128 blocks.
// 256 threads: thread = (p, q) owning 16 states (p fixed, n in [16q,16q+16)).
// Timesteps staged into LDS in chunks of 64 via async-to-LDS copies.
// ---------------------------------------------------------------------------
__global__ __launch_bounds__(256) void scan_kernel(
    const float* __restrict__ xBC, const float* __restrict__ zxBCdt,
    const float* __restrict__ dt_bias, const float* __restrict__ A_log,
    float* __restrict__ scan_out)
{
    const int db  = blockIdx.x >> 5;   // 0..3
    const int h   = blockIdx.x & 31;
    const int dir = db >> 1;           // 0 fwd, 1 bwd
    const int b   = db & 1;
    const int tid = threadIdx.x;
    const int p   = tid >> 2;
    const int q   = tid & 3;
    const int n0  = q << 4;

    __shared__ __align__(16) float sx[64 * 64];
    __shared__ __align__(16) float sB[64 * 64];
    __shared__ __align__(16) float sC[64 * 64];
    __shared__ float sdt[64];
    __shared__ float sdec[64];

    float st[16];
#pragma unroll
    for (int k = 0; k < 16; ++k) st[k] = 0.0f;

    const float Aval = -__expf(A_log[h]);
    const float dtb  = dt_bias[h];

    for (int t0 = 0; t0 < LQ; t0 += 64) {
        __syncthreads();
        // stage 64 steps of x/B/C (4096 floats each) as 16B async copies
#pragma unroll
        for (int rep = 0; rep < 4; ++rep) {
            int v = rep * 256 + tid;       // float4 index, 0..1023
            int i = v >> 4, e = (v & 15) << 2;
            int tt = dir ? (LQ - 1 - (t0 + i)) : (t0 + i);
            const float* row = xBC + (size_t)(b * LQ + tt) * DXBC;
            async_load16(&sx[i * 64 + e], row + h * HEADDIM + e);
            async_load16(&sB[i * 64 + e], row + DINNER + dir * 128 + e);
            async_load16(&sC[i * 64 + e], row + DINNER + dir * 128 + 64 + e);
        }
        if (tid < 64) {
            int i = tid;
            int tt = dir ? (LQ - 1 - (t0 + i)) : (t0 + i);
            float raw = zxBCdt[(size_t)(b * LQ + tt) * PITCH1 + (DINNER + DXBC) + dir * 32 + h] + dtb;
            float dte = (raw > 20.0f) ? raw : log1pf(__expf(raw));
            sdt[i]  = dte;
            sdec[i] = __expf(dte * Aval);
        }
        async_wait0();
        __syncthreads();

        for (int i = 0; i < 64; ++i) {
            float dte  = sdt[i];
            float dec  = sdec[i];
            float coef = dte * sx[i * 64 + p];
            const float4* Bv = (const float4*)&sB[i * 64 + n0];
            const float4* Cv = (const float4*)&sC[i * 64 + n0];
            float yp = 0.0f;
#pragma unroll
            for (int v = 0; v < 4; ++v) {
                float4 Bq = Bv[v], Cq = Cv[v];
                st[v*4+0] = st[v*4+0] * dec + coef * Bq.x; yp += st[v*4+0] * Cq.x;
                st[v*4+1] = st[v*4+1] * dec + coef * Bq.y; yp += st[v*4+1] * Cq.y;
                st[v*4+2] = st[v*4+2] * dec + coef * Bq.z; yp += st[v*4+2] * Cq.z;
                st[v*4+3] = st[v*4+3] * dec + coef * Bq.w; yp += st[v*4+3] * Cq.w;
            }
            yp += __shfl_xor(yp, 1, 32);
            yp += __shfl_xor(yp, 2, 32);
            if (q == 0)
                scan_out[((size_t)db * LQ + (t0 + i)) * DINNER + h * HEADDIM + p] = yp;
        }
    }
}

// ---------------------------------------------------------------------------
// Combine fwd/bwd (with 1-step shift), D-skip GEMV, RMS norm, SiLU(z) gate.
// One block per (b,t); writes bf16 gated activations for the out-proj GEMM.
// ---------------------------------------------------------------------------
__global__ __launch_bounds__(256) void combine_kernel(
    const float* __restrict__ xBC, const float* __restrict__ zxBCdt,
    const float* __restrict__ scan_out,
    const float* __restrict__ D_w, const float* __restrict__ D_b,
    const float* __restrict__ rms_w, unsigned short* __restrict__ ygate)
{
    const int t = blockIdx.x;
    const int b = blockIdx.y;
    const int tid = threadIdx.x;
    const size_t m = (size_t)b * LQ + t;

    __shared__ __align__(16) float sx[DINNER];
    __shared__ float sdk[NHEADS];
    __shared__ float sred[8];
    __shared__ float srstd;

    float y[8];
#pragma unroll
    for (int j = 0; j < 8; ++j) {
        int c = j * 256 + tid;
        sx[c] = xBC[m * DXBC + c];
        float yf = (t >= 1)
            ? scan_out[((size_t)b * LQ + (t - 1)) * DINNER + c] : 0.0f;
        float yb = (t <= LQ - 2)
            ? scan_out[((size_t)(2 + b) * LQ + (LQ - 2 - t)) * DINNER + c] : 0.0f;
        y[j] = yf + yb;
    }
    __syncthreads();

    // D-skip: dskip[h] = dot(x, D_w[h,:]) + D_b[h]; 8 lanes per head
    {
        int hh = tid >> 3, s = tid & 7;
        float acc = 0.0f;
        int c0 = s * 256;
        for (int i = 0; i < 256; ++i)
            acc += sx[c0 + i] * D_w[hh * DINNER + c0 + i];
        acc += __shfl_xor(acc, 1, 32);
        acc += __shfl_xor(acc, 2, 32);
        acc += __shfl_xor(acc, 4, 32);
        if (s == 0) sdk[hh] = acc + D_b[hh];
    }
    __syncthreads();

    float ss = 0.0f;
#pragma unroll
    for (int j = 0; j < 8; ++j) {
        int c = j * 256 + tid;
        y[j] += sdk[c >> 6] * sx[c];
        ss += y[j] * y[j];
    }
#pragma unroll
    for (int off = 1; off < 32; off <<= 1) ss += __shfl_xor(ss, off, 32);
    if ((tid & 31) == 0) sred[tid >> 5] = ss;
    __syncthreads();
    if (tid == 0) {
        float tot = 0.0f;
#pragma unroll
        for (int w = 0; w < 8; ++w) tot += sred[w];
        srstd = rsqrtf(tot / (float)DINNER + 1e-5f);
    }
    __syncthreads();
    const float rstd = srstd;

#pragma unroll
    for (int j = 0; j < 8; ++j) {
        int c = j * 256 + tid;
        float z = zxBCdt[m * PITCH1 + c];
        float val = y[j] * rstd * rms_w[c] * silu_f(z);
        ygate[m * DINNER + c] = bf16_rne(val);
    }
}

// ---------------------------------------------------------------------------
// Host launcher
// ---------------------------------------------------------------------------
extern "C" void kernel_launch(void* const* d_in, const int* in_sizes, int n_in,
                              void* d_out, int out_size, void* d_ws, size_t ws_size,
                              hipStream_t stream)
{
    const float* u         = (const float*)d_in[0];
    const float* in_proj_w = (const float*)d_in[1];
    const float* conv_w    = (const float*)d_in[2];
    const float* conv_b    = (const float*)d_in[3];
    const float* dt_bias   = (const float*)d_in[4];
    const float* A_log     = (const float*)d_in[5];
    const float* D_w       = (const float*)d_in[6];
    const float* D_b       = (const float*)d_in[7];
    const float* rms_w     = (const float*)d_in[8];
    const float* outproj_w = (const float*)d_in[9];
    float* out = (float*)d_out;

    char* base = (char*)d_ws;
    size_t off = 0;
    auto take = [&](size_t bytes) -> void* {
        void* p = base + off;
        off += (bytes + 255) & ~(size_t)255;
        return p;
    };
    float* zxBCdt = (float*)take((size_t)MROWS * PITCH1 * 4);
    float* xBC    = (float*)take((size_t)MROWS * DXBC * 4);
    float* scan_o = (float*)take((size_t)4 * LQ * DINNER * 4);
    unsigned short* u_bf  = (unsigned short*)take((size_t)MROWS * DMODEL * 2);
    unsigned short* w1_bf = (unsigned short*)take((size_t)PITCH1 * DMODEL * 2); // padded rows
    unsigned short* w2_bf = (unsigned short*)take((size_t)DMODEL * DINNER * 2);
    unsigned short* yg_bf = (unsigned short*)take((size_t)MROWS * DINNER * 2);

    // bf16 conversions
    {
        int n = MROWS * DMODEL;
        cvt_bf16_kernel<<<(n + 255) / 256, 256, 0, stream>>>(u, u_bf, n);
    }
    {
        int n = DINPROJ * DMODEL;   // pad rows of w1_bf are never consumed
        cvt_bf16_kernel<<<(n + 255) / 256, 256, 0, stream>>>(in_proj_w, w1_bf, n);
    }
    {
        int n = DMODEL * DINNER;
        cvt_bf16_kernel<<<(n + 255) / 256, 256, 0, stream>>>(outproj_w, w2_bf, n);
    }

    // in_proj: [8192,1024] x [4480,1024]^T -> [8192,4480] (cols >=4416 unused)
    gemm_bf16_wmma<<<dim3(PITCH1 / 128, MROWS / 256), 256, 0, stream>>>(
        (const __bf16*)u_bf, (const __bf16*)w1_bf, zxBCdt, DMODEL, PITCH1);

    // depthwise conv + silu
    {
        size_t n = (size_t)MROWS * DXBC;
        conv_silu_kernel<<<(unsigned)(n / 256), 256, 0, stream>>>(
            zxBCdt, conv_w, conv_b, xBC);
    }

    // bidirectional selective scan
    scan_kernel<<<128, 256, 0, stream>>>(xBC, zxBCdt, dt_bias, A_log, scan_o);

    // combine + D-skip + RMS norm + gate -> bf16
    combine_kernel<<<dim3(LQ, 2), 256, 0, stream>>>(
        xBC, zxBCdt, scan_o, D_w, D_b, rms_w, yg_bf);

    // out_proj: [8192,2048] x [1024,2048]^T -> [8192,1024]
    gemm_bf16_wmma<<<dim3(DMODEL / 128, MROWS / 256), 256, 0, stream>>>(
        (const __bf16*)yg_bf, (const __bf16*)w2_bf, out, DINNER, DMODEL);
}
